// NAFAlignBlock_74019466379845
// MI455X (gfx1250) — matrix operand order repaired and checked by reading, not compiled, not run on hardware
//
#include <hip/hip_runtime.h>

// ---------------- problem constants ----------------
#define B_   2
#define C0_  64
#define H_   160
#define W_   160
#define DW_  128
#define G_   4
#define OM_  112
#define HW_  (H_*W_)
#define P_   (B_*HW_)        // 51200 pixels

typedef __attribute__((ext_vector_type(16))) _Float16 v16h;
typedef __attribute__((ext_vector_type(8)))  _Float16 v8h;
typedef __attribute__((ext_vector_type(8)))  float    v8f;

union AFrag { v16h v; v8h h[2]; };

__device__ __forceinline__ void async_wait0() {
#if __has_builtin(__builtin_amdgcn_s_wait_asynccnt)
  __builtin_amdgcn_s_wait_asynccnt(0);
#else
  asm volatile("s_wait_asynccnt 0x0" ::: "memory");
#endif
}

// ---------------- weight f32 -> f16 ----------------
__global__ __launch_bounds__(256) void k_f32_to_f16(const float* __restrict__ s,
                                                    _Float16* __restrict__ d, int n) {
  int i = blockIdx.x * 256 + threadIdx.x;
  if (i < n) d[i] = (_Float16)s[i];
}

// ---------------- WMMA GEMM: C[M,N] = A[M,K] * Wt[N,K]^T + bias ----------------
// A f16 row-major, Wt f16 row-major (N rows of K), C f32 or f16 row-major.
// Wave tile 32(M) x 16*NSUB(N); block = 8 waves -> 256 rows. M % 256 == 0,
// K % 32 == 0. Weight tile (16*NSUB x K, contiguous rows) staged to LDS via
// async global->LDS DMA; B fragments then come from ds_load.
template<int NSUB, bool OUT_F16>
__global__ __launch_bounds__(256) void k_gemm_wmma(
    const _Float16* __restrict__ A,
    const _Float16* __restrict__ Wt,
    const float* __restrict__ bias,
    void* __restrict__ Cout,
    int M, int N, int K, int n_base)
{
  __shared__ _Float16 smem[64 * 128];          // up to 64 rows x K=128, 16 KB

  const int lane = threadIdx.x & 31;
  const int wave = threadIdx.x >> 5;
  const int m0   = blockIdx.x * 256 + wave * 32;
  const int n0   = n_base + blockIdx.y * 64;

  // ---- stage Wt rows [n0, n0+16*NSUB) into LDS (contiguous region) ----
  {
    const int chunks = (16 * NSUB * K * 2) >> 4;     // 16-byte chunks
    const _Float16* gsrc = Wt + (size_t)n0 * K;
    const unsigned lds0 = (unsigned)(uintptr_t)smem; // low 32 bits = LDS offset
    for (int i = threadIdx.x; i < chunks; i += 256) {
      unsigned ldsoff = lds0 + i * 16;
      const _Float16* ga = gsrc + i * 8;
      asm volatile("global_load_async_to_lds_b128 %0, %1, off"
                   :: "v"(ldsoff), "v"(ga) : "memory");
    }
    async_wait0();
    __syncthreads();
  }

  const int half = lane >> 4;       // 0: low half-lanes, 1: high
  const int l15  = lane & 15;

  v8f acc[2][NSUB] = {};

  // A fragment: lane<16 -> M=lane, K {k0..k0+15};
  // lane>=16 -> same M, K {k0+16..k0+31}.
  const _Float16* arow0 = A + (size_t)(m0 + l15) * K + half * 16;
  const _Float16* arow1 = arow0 + (size_t)16 * K;

  for (int k0 = 0; k0 < K; k0 += 32) {
    AFrag a0, a1;
    a0.h[0] = *(const v8h*)(arow0 + k0);
    a0.h[1] = *(const v8h*)(arow0 + k0 + 8);
    a1.h[0] = *(const v8h*)(arow1 + k0);
    a1.h[1] = *(const v8h*)(arow1 + k0 + 8);
    // B fragment: lane<16 -> N=n, K {k0..k0+15}; lane>=16 -> K {k0+16..k0+31}
    #pragma unroll
    for (int t = 0; t < NSUB; ++t) {
      v16h b = *(const v16h*)(smem + (size_t)(t * 16 + l15) * K + k0 + half * 16);
      acc[0][t] = __builtin_amdgcn_wmma_f32_16x16x32_f16(
                    false, a0.v, false, b, (short)0, acc[0][t], false, false);
      acc[1][t] = __builtin_amdgcn_wmma_f32_16x16x32_f16(
                    false, a1.v, false, b, (short)0, acc[1][t], false, false);
    }
  }

  // C layout: VGPR r -> (M = m0 + mi*16 + half*8 + r, N = n0 + t*16 + l15)
  #pragma unroll
  for (int mi = 0; mi < 2; ++mi) {
    const int row0 = m0 + mi * 16 + half * 8;
    #pragma unroll
    for (int t = 0; t < NSUB; ++t) {
      const int col = n0 + t * 16 + l15;
      const float bb = bias[col];
      #pragma unroll
      for (int r = 0; r < 8; ++r) {
        float vv = acc[mi][t][r] + bb;
        if constexpr (OUT_F16)
          ((_Float16*)Cout)[(size_t)(row0 + r) * N + col] = (_Float16)vv;
        else
          ((float*)Cout)[(size_t)(row0 + r) * N + col] = vv;
      }
    }
  }
}

template<bool OUT_F16>
static void launch_gemm(const _Float16* A, const _Float16* Wt, const float* bias,
                        void* C, int M, int N, int K, hipStream_t s) {
  const int gM    = M / 256;
  const int nfull = N / 64;
  const int tail  = (N % 64) / 16;
  if (nfull > 0)
    k_gemm_wmma<4, OUT_F16><<<dim3(gM, nfull), 256, 0, s>>>(A, Wt, bias, C, M, N, K, 0);
  if (tail == 3)
    k_gemm_wmma<3, OUT_F16><<<dim3(gM, 1), 256, 0, s>>>(A, Wt, bias, C, M, N, K, nfull * 64);
  else if (tail == 2)
    k_gemm_wmma<2, OUT_F16><<<dim3(gM, 1), 256, 0, s>>>(A, Wt, bias, C, M, N, K, nfull * 64);
  else if (tail == 1)
    k_gemm_wmma<1, OUT_F16><<<dim3(gM, 1), 256, 0, s>>>(A, Wt, bias, C, M, N, K, nfull * 64);
}

// ---------------- LayerNorm over 64 channels (NCHW in, f16 NHWC out) ----------------
__global__ __launch_bounds__(256) void k_ln64(const float* __restrict__ x,
                                              const float* __restrict__ g,
                                              _Float16* __restrict__ out) {
  int p = blockIdx.x * 256 + threadIdx.x;
  if (p >= P_) return;
  int b = p / HW_, hw = p - b * HW_;
  const float* px = x + (size_t)b * C0_ * HW_ + hw;
  float v[C0_];
  float s = 0.f;
  #pragma unroll
  for (int c = 0; c < C0_; ++c) { v[c] = px[(size_t)c * HW_]; s += v[c]; }
  float m = s * (1.f / C0_);
  float vs = 0.f;
  #pragma unroll
  for (int c = 0; c < C0_; ++c) { float d = v[c] - m; vs += d * d; }
  float rs = rsqrtf(vs * (1.f / C0_) + 1e-5f);
  #pragma unroll
  for (int c = 0; c < C0_; ++c)
    out[(size_t)p * C0_ + c] = (_Float16)((v[c] - m) * rs * g[c]);
}

// ---------------- depthwise 3x3 SAME on f16 NHWC (+bias), f16 out ----------------
__global__ __launch_bounds__(256) void k_dwc(const _Float16* __restrict__ h,
                                             const float* __restrict__ w,
                                             const float* __restrict__ bias,
                                             _Float16* __restrict__ out) {
  int idx = blockIdx.x * 256 + threadIdx.x;           // P_*DW_
  if (idx >= P_ * DW_) return;
  int c = idx & (DW_ - 1);
  int p = idx >> 7;
  int b = p / HW_, rem = p - b * HW_;
  int y = rem / W_, x = rem - y * W_;
  float acc = bias[c];
  #pragma unroll
  for (int dy = 0; dy < 3; ++dy) {
    int yy = y + dy - 1;
    if (yy < 0 || yy >= H_) continue;
    #pragma unroll
    for (int dx = 0; dx < 3; ++dx) {
      int xx = x + dx - 1;
      if (xx < 0 || xx >= W_) continue;
      acc += (float)h[(((size_t)b * H_ + yy) * W_ + xx) * DW_ + c] * w[(dy * 3 + dx) * DW_ + c];
    }
  }
  out[idx] = (_Float16)acc;
}

// ---------------- deformable bilinear sampling ----------------
// v: f32 [B,H,W,128]; om: f32 [P,112] (g*27 + k*3 + {offx,offy,mask})
// thread = (pixel, group, 8-channel slice); out f16 [P,128]
__global__ __launch_bounds__(256) void k_dcn(const float* __restrict__ v,
                                             const float* __restrict__ om,
                                             _Float16* __restrict__ out) {
  int tid = blockIdx.x * 256 + threadIdx.x;           // P_*16
  if (tid >= P_ * 16) return;
  int c8 = tid & 3;
  int g  = (tid >> 2) & 3;
  int p  = tid >> 4;
  int b = p / HW_, rem = p - b * HW_;
  int y = rem / W_, x = rem - y * W_;
  const int cbase = g * 32 + c8 * 8;
  float acc[8] = {0,0,0,0,0,0,0,0};
  const float* omg = om + (size_t)p * OM_ + g * 27;
  #pragma unroll
  for (int k = 0; k < 9; ++k) {
    float offx = omg[k * 3 + 0], offy = omg[k * 3 + 1], mask = omg[k * 3 + 2];
    // padded-frame coords: px = x + 1.5 + (k%3-1) + offx - 0.5 = x + k%3 + offx
    float pxc = (float)(x + (k % 3)) + offx;
    float pyc = (float)(y + (k / 3)) + offy;
    float x0f = floorf(pxc), y0f = floorf(pyc);
    float tx = pxc - x0f, ty = pyc - y0f;
    int x0 = (int)x0f, y0 = (int)y0f;
    #pragma unroll
    for (int corner = 0; corner < 4; ++corner) {
      int dx = corner & 1, dy = corner >> 1;
      int xi = x0 + dx, yi = y0 + dy;
      // padded frame is [0, W+1]; only interior maps to data, ring/outside contribute 0
      if (xi < 1 || xi > W_ || yi < 1 || yi > H_) continue;
      float wgt = (dx ? tx : 1.f - tx) * (dy ? ty : 1.f - ty) * mask;
      const float* src = v + (((size_t)b * H_ + (yi - 1)) * W_ + (xi - 1)) * DW_ + cbase;
      #pragma unroll
      for (int j = 0; j < 8; ++j) acc[j] += wgt * src[j];
    }
  }
  _Float16* dst = out + (size_t)p * DW_ + cbase;
  #pragma unroll
  for (int j = 0; j < 8; ++j) dst[j] = (_Float16)acc[j];
}

// ---------------- SimpleGate: [P,128] -> [P,64], f32 or f16 out ----------------
template<bool OUT_F16>
__global__ __launch_bounds__(256) void k_sg(const float* __restrict__ in, void* __restrict__ out) {
  int idx = blockIdx.x * 256 + threadIdx.x;           // P_*64
  if (idx >= P_ * 64) return;
  int c = idx & 63, p = idx >> 6;
  float r = in[(size_t)p * 128 + c] * in[(size_t)p * 128 + 64 + c];
  if constexpr (OUT_F16) ((_Float16*)out)[idx] = (_Float16)r;
  else                   ((float*)out)[idx] = r;
}

// ---------------- pooled mean over HW per (b, concat-channel) ----------------
__global__ __launch_bounds__(256) void k_pool(const float* __restrict__ fL,
                                              const float* __restrict__ fR,
                                              float* __restrict__ pooled) {
  __shared__ float red[256];
  int bc = blockIdx.x;                 // b*128 + c
  int b = bc >> 7, c = bc & 127;
  const float* src = (c < 64) ? fL : fR;
  int ch = c & 63;
  float s = 0.f;
  for (int i = threadIdx.x; i < HW_; i += 256)
    s += src[((size_t)b * HW_ + i) * 64 + ch];
  red[threadIdx.x] = s;
  __syncthreads();
  for (int st = 128; st > 0; st >>= 1) {
    if (threadIdx.x < st) red[threadIdx.x] += red[threadIdx.x + st];
    __syncthreads();
  }
  if (threadIdx.x == 0) pooled[bc] = red[0] * (1.f / HW_);
}

// ---------------- scale[b,o] = sca_w[o,:] . pooled[b,:] + sca_b[o] ----------------
__global__ __launch_bounds__(256) void k_scale(const float* __restrict__ pooled,
                                               const float* __restrict__ sw,
                                               const float* __restrict__ sb,
                                               float* __restrict__ scale) {
  int t = threadIdx.x;                 // 256 = B_*128
  int b = t >> 7, o = t & 127;
  float s = sb[o];
  for (int c = 0; c < 128; ++c) s += sw[o * 128 + c] * pooled[b * 128 + c];
  scale[t] = s;
}

// ---------------- xs[p,c] = feat * scale, f16 out ----------------
__global__ __launch_bounds__(256) void k_scale_apply(const float* __restrict__ fL,
                                                     const float* __restrict__ fR,
                                                     const float* __restrict__ scale,
                                                     _Float16* __restrict__ xs) {
  int idx = blockIdx.x * 256 + threadIdx.x;           // P_*128
  if (idx >= P_ * 128) return;
  int c = idx & 127, p = idx >> 7;
  int b = p / HW_;
  float f = (c < 64) ? fL[(size_t)p * 64 + c] : fR[(size_t)p * 64 + (c - 64)];
  xs[idx] = (_Float16)(f * scale[b * 128 + c]);
}

// ---------------- y = x + xc*beta (NCHW x in, NHWC y out, both sides) ----------------
__global__ __launch_bounds__(256) void k_make_y(const float* __restrict__ xc,
                                                const float* __restrict__ xl,
                                                const float* __restrict__ xr,
                                                const float* __restrict__ beta,
                                                float* __restrict__ yL, float* __restrict__ yR) {
  int idx = blockIdx.x * 256 + threadIdx.x;           // P_*64
  if (idx >= P_ * 64) return;
  int c = idx & 63, p = idx >> 6;
  int b = p / HW_, hw = p - b * HW_;
  float t = xc[idx] * beta[c];
  size_t nchw = ((size_t)b * C0_ + c) * HW_ + hw;
  yL[idx] = xl[nchw] + t;
  yR[idx] = xr[nchw] + t;
}

// ---------------- LayerNorm over concat-128 channels, f16 out ----------------
__global__ __launch_bounds__(256) void k_ln128(const float* __restrict__ yL,
                                               const float* __restrict__ yR,
                                               const float* __restrict__ g,
                                               _Float16* __restrict__ zin) {
  int p = blockIdx.x * 256 + threadIdx.x;
  if (p >= P_) return;
  const float4* a  = (const float4*)(yL + (size_t)p * 64);
  const float4* b4 = (const float4*)(yR + (size_t)p * 64);
  float s = 0.f, ss = 0.f;
  #pragma unroll
  for (int i = 0; i < 16; ++i) {
    float4 u = a[i], w = b4[i];
    s  += u.x + u.y + u.z + u.w + w.x + w.y + w.z + w.w;
    ss += u.x*u.x + u.y*u.y + u.z*u.z + u.w*u.w + w.x*w.x + w.y*w.y + w.z*w.z + w.w*w.w;
  }
  float m  = s * (1.f / 128.f);
  float rs = rsqrtf(ss * (1.f / 128.f) - m * m + 1e-5f);
  _Float16* zp = zin + (size_t)p * 128;
  #pragma unroll
  for (int i = 0; i < 16; ++i) {
    float4 u = a[i];
    zp[i*4+0] = (_Float16)((u.x - m) * rs * g[i*4+0]);
    zp[i*4+1] = (_Float16)((u.y - m) * rs * g[i*4+1]);
    zp[i*4+2] = (_Float16)((u.z - m) * rs * g[i*4+2]);
    zp[i*4+3] = (_Float16)((u.w - m) * rs * g[i*4+3]);
  }
  #pragma unroll
  for (int i = 0; i < 16; ++i) {
    float4 w = b4[i];
    zp[64+i*4+0] = (_Float16)((w.x - m) * rs * g[64+i*4+0]);
    zp[64+i*4+1] = (_Float16)((w.y - m) * rs * g[64+i*4+1]);
    zp[64+i*4+2] = (_Float16)((w.z - m) * rs * g[64+i*4+2]);
    zp[64+i*4+3] = (_Float16)((w.w - m) * rs * g[64+i*4+3]);
  }
}

// ---------------- out = y + z*gamma, NHWC -> NCHW ----------------
__global__ __launch_bounds__(256) void k_final(const float* __restrict__ yL,
                                               const float* __restrict__ yR,
                                               const float* __restrict__ z,
                                               const float* __restrict__ gamma,
                                               float* __restrict__ outL, float* __restrict__ outR) {
  int idx = blockIdx.x * 256 + threadIdx.x;           // P_*64
  if (idx >= P_ * 64) return;
  int c = idx & 63, p = idx >> 6;
  int b = p / HW_, hw = p - b * HW_;
  float zg = z[idx] * gamma[c];
  size_t nchw = ((size_t)b * C0_ + c) * HW_ + hw;
  outL[nchw] = yL[idx] + zg;
  outR[nchw] = yR[idx] + zg;
}

// ---------------- host launch ----------------
extern "C" void kernel_launch(void* const* d_in, const int* in_sizes, int n_in,
                              void* d_out, int out_size, void* d_ws, size_t ws_size,
                              hipStream_t stream) {
  const float* x_l     = (const float*)d_in[0];
  const float* x_r     = (const float*)d_in[1];
  const float* ln1_g   = (const float*)d_in[2];
  const float* pw1_w   = (const float*)d_in[3];
  const float* pw1_b   = (const float*)d_in[4];
  const float* val_w   = (const float*)d_in[5];
  const float* val_b   = (const float*)d_in[6];
  const float* dwc_w   = (const float*)d_in[7];
  const float* dwc_b   = (const float*)d_in[8];
  const float* om_w    = (const float*)d_in[9];
  const float* om_b    = (const float*)d_in[10];
  const float* outp_w  = (const float*)d_in[11];
  const float* outp_b  = (const float*)d_in[12];
  const float* sca_w   = (const float*)d_in[13];
  const float* sca_b   = (const float*)d_in[14];
  const float* conv3_w = (const float*)d_in[15];
  const float* conv3_b = (const float*)d_in[16];
  const float* norm2_g = (const float*)d_in[17];
  const float* conv4_w = (const float*)d_in[18];
  const float* conv4_b = (const float*)d_in[19];
  const float* conv5_w = (const float*)d_in[20];
  const float* conv5_b = (const float*)d_in[21];
  const float* beta    = (const float*)d_in[22];
  const float* gamma   = (const float*)d_in[23];
  (void)in_sizes; (void)n_in; (void)out_size; (void)ws_size;

  char* ws = (char*)d_ws;
  size_t off = 0;
  auto alloc = [&](size_t bytes) -> char* {
    char* pp = ws + off;
    off = (off + bytes + 255) & ~(size_t)255;
    return pp;
  };

  // f16 weights
  _Float16* pw1h  = (_Float16*)alloc(128 * 64 * 2);
  _Float16* valh  = (_Float16*)alloc(128 * 128 * 2);
  _Float16* omh   = (_Float16*)alloc(112 * 128 * 2);
  _Float16* outph = (_Float16*)alloc(128 * 128 * 2);
  _Float16* c3h   = (_Float16*)alloc(64 * 128 * 2);
  _Float16* c4h   = (_Float16*)alloc(128 * 128 * 2);
  _Float16* c5h   = (_Float16*)alloc(64 * 64 * 2);
  float* pooled   = (float*)alloc(256 * 4);
  float* scal     = (float*)alloc(256 * 4);
  // persistent activations
  float* featL = (float*)alloc((size_t)P_ * 64 * 4);
  float* featR = (float*)alloc((size_t)P_ * 64 * 4);
  float* yL    = (float*)alloc((size_t)P_ * 64 * 4);
  float* yR    = (float*)alloc((size_t)P_ * 64 * 4);
  // reusable temporaries
  _Float16* tA  = (_Float16*)alloc((size_t)P_ * 64 * 2);   // ln64 out / z2
  _Float16* tH  = (_Float16*)alloc((size_t)P_ * 128 * 2);  // h / dcnout / xs / zin
  float*    tV  = (float*)alloc((size_t)P_ * 128 * 4);     // v / g4out / z1
  _Float16* tDW = (_Float16*)alloc((size_t)P_ * 128 * 2);  // dw ; aliased as f32 [P,64] xc
  float*    tOM = (float*)alloc((size_t)P_ * 112 * 4);     // om ; later z [P,64]

  auto cvt = [&](const float* s, _Float16* d, int n) {
    k_f32_to_f16<<<(n + 255) / 256, 256, 0, stream>>>(s, d, n);
  };
  cvt(pw1_w,   pw1h,  128 * 64);
  cvt(val_w,   valh,  128 * 128);
  cvt(om_w,    omh,   112 * 128);
  cvt(outp_w,  outph, 128 * 128);
  cvt(conv3_w, c3h,   64 * 128);
  cvt(conv4_w, c4h,   128 * 128);
  cvt(conv5_w, c5h,   64 * 64);

  dim3 blk(256);

  for (int side = 0; side < 2; ++side) {
    const float* xin = side ? x_r : x_l;
    float* feat      = side ? featR : featL;
    k_ln64<<<(P_ + 255) / 256, blk, 0, stream>>>(xin, ln1_g, tA);
    launch_gemm<true >(tA,  pw1h,  pw1_b,  tH,  P_, 128, 64,  stream);
    launch_gemm<false>(tH,  valh,  val_b,  tV,  P_, 128, 128, stream);
    k_dwc<<<(P_ * 128 + 255) / 256, blk, 0, stream>>>(tH, dwc_w, dwc_b, tDW);
    launch_gemm<false>(tDW, omh,   om_b,   tOM, P_, 112, 128, stream);
    k_dcn<<<(P_ * 16 + 255) / 256, blk, 0, stream>>>(tV, tOM, tH);
    launch_gemm<false>(tH,  outph, outp_b, tV,  P_, 128, 128, stream);
    k_sg<false><<<(P_ * 64 + 255) / 256, blk, 0, stream>>>(tV, feat);
  }

  k_pool<<<256, blk, 0, stream>>>(featL, featR, pooled);
  k_scale<<<1, blk, 0, stream>>>(pooled, sca_w, sca_b, scal);
  k_scale_apply<<<(P_ * 128 + 255) / 256, blk, 0, stream>>>(featL, featR, scal, tH);
  launch_gemm<false>(tH, c3h, conv3_b, (float*)tDW, P_, 64, 128, stream);
  k_make_y<<<(P_ * 64 + 255) / 256, blk, 0, stream>>>((const float*)tDW, x_l, x_r, beta, yL, yR);
  k_ln128<<<(P_ + 255) / 256, blk, 0, stream>>>(yL, yR, norm2_g, tH);
  launch_gemm<false>(tH, c4h, conv4_b, tV, P_, 128, 128, stream);
  k_sg<true><<<(P_ * 64 + 255) / 256, blk, 0, stream>>>(tV, tA);
  launch_gemm<false>(tA, c5h, conv5_b, tOM, P_, 64, 64, stream);

  float* outL = (float*)d_out;
  float* outR = outL + (size_t)B_ * C0_ * HW_;
  k_final<<<(P_ * 64 + 255) / 256, blk, 0, stream>>>(yL, yR, tOM, gamma, outL, outR);
}
